// MultiHead_86062554677874
// MI455X (gfx1250) — matrix operand order repaired and checked by reading, not compiled, not run on hardware
//
#include <hip/hip_runtime.h>

// ---------------------------------------------------------------------------
// Cosine-similarity GEMM for MI455X (gfx1250, wave32, WMMA bf16 path)
//   out[b,o] = <x[b], w[o]> / (max(||x[b]||,eps) * max(||w[o]||,eps))
//   M=4096, N=8192, K=4096.  275 GFLOP vs 320 MB -> compute-bound ->
//   v_wmma_f32_16x16x32_bf16 with f32 accumulation.
//
// Fast path (ws >= 96MB): fused f32->bf16 conversion + row-norm pass into
// d_ws; GEMM streams bf16 (L2-resident: 96MB < 192MB L2) with a double-
// buffered LDS pipeline; next-tile global loads pinned ABOVE the WMMA block
// via sched_barrier so their latency hides under the matrix ops.
// Fallback (small ws): fused-conversion GEMM.
// ---------------------------------------------------------------------------

typedef __attribute__((ext_vector_type(16))) __bf16 v16bf;
typedef __attribute__((ext_vector_type(8)))  __bf16 v8bf;
typedef __attribute__((ext_vector_type(2)))  __bf16 v2bf;
typedef __attribute__((ext_vector_type(8)))  float  v8f;

#define M_DIM 4096
#define N_DIM 8192
#define K_DIM 4096
#define EPS_F 1e-8f

#define BM 256
#define BN 128
#define BK 32
#define LDK 40                      // padded LDS row stride (bf16 elems)
#define A_ELEMS (BM * LDK)          // 10240
#define HALF_ELEMS (A_ELEMS + BN * LDK)   // 15360 elems = 30720 B per buffer

#if __has_builtin(__builtin_amdgcn_sched_barrier)
#define SCHED_FENCE() __builtin_amdgcn_sched_barrier(0)
#else
#define SCHED_FENCE() do {} while (0)
#endif

// ---- f32 -> bf16 (RNE) -----------------------------------------------------
__device__ __forceinline__ unsigned short f2bf_sw(float f) {
  unsigned int u = __builtin_bit_cast(unsigned int, f);
  u += 0x7FFFu + ((u >> 16) & 1u);
  return (unsigned short)(u >> 16);
}
__device__ __forceinline__ uint2 pk4_bf16(float4 v) {
#if __has_builtin(__builtin_amdgcn_cvt_pk_bf16_f32)
  v2bf lo = __builtin_amdgcn_cvt_pk_bf16_f32(v.x, v.y);
  v2bf hi = __builtin_amdgcn_cvt_pk_bf16_f32(v.z, v.w);
  return uint2{ __builtin_bit_cast(unsigned int, lo),
                __builtin_bit_cast(unsigned int, hi) };
#else
  return uint2{ (unsigned int)f2bf_sw(v.x) | ((unsigned int)f2bf_sw(v.y) << 16),
                (unsigned int)f2bf_sw(v.z) | ((unsigned int)f2bf_sw(v.w) << 16) };
#endif
}

// ---- fragment assembly (CDNA5 16-bit WMMA layouts) -------------------------
__device__ __forceinline__ v16bf frag16(const unsigned short* p, int gap) {
  v8bf lo = *(const v8bf*)p;
  v8bf hi = *(const v8bf*)(p + gap);
  return __builtin_shufflevector(lo, hi,
      0, 1, 2, 3, 4, 5, 6, 7, 8, 9, 10, 11, 12, 13, 14, 15);
}

// ---------------------------------------------------------------------------
// Fused convert + inverse-norm pass (fast path), one block per row.
// ---------------------------------------------------------------------------
__global__ __launch_bounds__(256) void conv_norm_kernel(
    const float* __restrict__ src, unsigned short* __restrict__ dst,
    float* __restrict__ inv, int K) {
  const int row = blockIdx.x;
  const float* p = src + (size_t)row * K;
  unsigned short* q = dst + (size_t)row * K;
  float s = 0.f;
  for (int i = threadIdx.x; i < (K >> 2); i += 256) {
    float4 v = *(const float4*)(p + i * 4);
    s += v.x * v.x + v.y * v.y + v.z * v.z + v.w * v.w;
    *(uint2*)(q + i * 4) = pk4_bf16(v);
  }
  #pragma unroll
  for (int off = 16; off > 0; off >>= 1) s += __shfl_down(s, off, 32);
  __shared__ float ws[8];
  const int lane = threadIdx.x & 31, w = threadIdx.x >> 5;
  if (lane == 0) ws[w] = s;
  __syncthreads();
  if (threadIdx.x == 0) {
    float t = 0.f;
    #pragma unroll
    for (int i = 0; i < 8; ++i) t += ws[i];
    inv[row] = 1.0f / fmaxf(sqrtf(t), EPS_F);
  }
}

// Standalone inverse-norm (fallback path)
__global__ __launch_bounds__(256) void inv_norm_kernel(
    const float* __restrict__ src, float* __restrict__ inv, int K) {
  const int row = blockIdx.x;
  const float4* p = (const float4*)(src + (size_t)row * K);
  float s = 0.f;
  for (int i = threadIdx.x; i < (K >> 2); i += 256) {
    float4 v = p[i];
    s += v.x * v.x + v.y * v.y + v.z * v.z + v.w * v.w;
  }
  #pragma unroll
  for (int off = 16; off > 0; off >>= 1) s += __shfl_down(s, off, 32);
  __shared__ float ws[8];
  const int lane = threadIdx.x & 31, w = threadIdx.x >> 5;
  if (lane == 0) ws[w] = s;
  __syncthreads();
  if (threadIdx.x == 0) {
    float t = 0.f;
    #pragma unroll
    for (int i = 0; i < 8; ++i) t += ws[i];
    inv[row] = 1.0f / fmaxf(sqrtf(t), EPS_F);
  }
}

// ---------------------------------------------------------------------------
// FAST PATH GEMM: bf16 operands, double-buffered LDS, 1 barrier per K-step,
// unconditional steady-state loads (peeled tail), loads pinned above WMMAs.
// Block tile 256x128, 8 waves 4(M)x2(N), each 64x64 via 4x4 WMMA accums.
// ---------------------------------------------------------------------------
__global__ __launch_bounds__(256, 2) void cos_gemm_bf16_kernel(
    const unsigned short* __restrict__ Ab,   // x  bf16 [M,K]
    const unsigned short* __restrict__ Wb,   // w  bf16 [N,K]
    const float* __restrict__ invx,
    const float* __restrict__ invw,
    float* __restrict__ out)
{
  __shared__ unsigned short lds[2 * HALF_ELEMS];   // 60 KB double buffer
  __shared__ float sIx[BM];
  __shared__ float sIw[BN];

  const int tid = threadIdx.x;
  const int m0 = blockIdx.y * BM;
  const int n0 = blockIdx.x * BN;

  sIx[tid] = invx[m0 + tid];
  if (tid < BN) sIw[tid] = invw[n0 + tid];

  const int wave  = tid >> 5;
  const int lane  = tid & 31;
  const int wm    = wave & 3;        // M offset wm*64
  const int wn    = wave >> 2;       // N offset wn*64
  const int lhalf = lane >> 4;
  const int l16   = lane & 15;

  // ---- staging geometry: 16B chunks of 8 bf16. A: 1024 chunks (4/thread),
  //      B: 512 chunks (2/thread). All offsets are named scalars.
  const int ar0 = tid >> 2;                  // rows 0..63
  const int ac  = (tid & 3) << 3;            // bf16 col 0/8/16/24
  const int ar1 = ar0 + 64, ar2 = ar0 + 128, ar3 = ar0 + 192;
  const int br0 = ar0, br1 = ar0 + 64;       // B rows 0..127

  const size_t ga0 = (size_t)(m0 + ar0) * K_DIM + ac;
  const size_t ga1 = (size_t)(m0 + ar1) * K_DIM + ac;
  const size_t ga2 = (size_t)(m0 + ar2) * K_DIM + ac;
  const size_t ga3 = (size_t)(m0 + ar3) * K_DIM + ac;
  const size_t gb0 = (size_t)(n0 + br0) * K_DIM + ac;
  const size_t gb1 = (size_t)(n0 + br1) * K_DIM + ac;

  const int sa0 = ar0 * LDK + ac, sa1 = ar1 * LDK + ac;
  const int sa2 = ar2 * LDK + ac, sa3 = ar3 * LDK + ac;
  const int sb0 = A_ELEMS + br0 * LDK + ac, sb1 = A_ELEMS + br1 * LDK + ac;

  // fragment base offsets within a buffer
  const int fa0 = (wm * 64 +  0 + l16) * LDK + (lhalf << 3);
  const int fa1 = (wm * 64 + 16 + l16) * LDK + (lhalf << 3);
  const int fa2 = (wm * 64 + 32 + l16) * LDK + (lhalf << 3);
  const int fa3 = (wm * 64 + 48 + l16) * LDK + (lhalf << 3);
  const int fb0 = A_ELEMS + (wn * 64 +  0 + l16) * LDK + (lhalf << 3);
  const int fb1 = A_ELEMS + (wn * 64 + 16 + l16) * LDK + (lhalf << 3);
  const int fb2 = A_ELEMS + (wn * 64 + 32 + l16) * LDK + (lhalf << 3);
  const int fb3 = A_ELEMS + (wn * 64 + 48 + l16) * LDK + (lhalf << 3);

  v8f acc[4][4];
  #pragma unroll
  for (int i = 0; i < 4; ++i)
    #pragma unroll
    for (int j = 0; j < 4; ++j) acc[i][j] = v8f{};

  uint4 p0, p1, p2, p3, p4, p5;   // named scalars -> register resident

#define LOAD_TILES(KK)                                            \
  do {                                                            \
    p0 = *(const uint4*)(Ab + ga0 + (KK));                        \
    p1 = *(const uint4*)(Ab + ga1 + (KK));                        \
    p2 = *(const uint4*)(Ab + ga2 + (KK));                        \
    p3 = *(const uint4*)(Ab + ga3 + (KK));                        \
    p4 = *(const uint4*)(Wb + gb0 + (KK));                        \
    p5 = *(const uint4*)(Wb + gb1 + (KK));                        \
  } while (0)

#define STORE_TILES(BUF)                                          \
  do {                                                            \
    *(uint4*)&lds[(BUF) + sa0] = p0;                              \
    *(uint4*)&lds[(BUF) + sa1] = p1;                              \
    *(uint4*)&lds[(BUF) + sa2] = p2;                              \
    *(uint4*)&lds[(BUF) + sa3] = p3;                              \
    *(uint4*)&lds[(BUF) + sb0] = p4;                              \
    *(uint4*)&lds[(BUF) + sb1] = p5;                              \
  } while (0)

// MORE is a literal 1/0 -> no runtime condition, no phi-merged loads.
// SCHED_FENCE pins the global loads above the DS/WMMA block so their
// latency is hidden under the 16 matrix ops.
#define GEMM_STEP(CUR, NXT, KNEXT, MORE)                          \
  do {                                                            \
    if (MORE) {                                                   \
      LOAD_TILES(KNEXT);                                          \
      SCHED_FENCE();                                              \
    }                                                             \
    const unsigned short* cb = &lds[(CUR)];                       \
    v16bf a0 = frag16(cb + fa0, 16), a1 = frag16(cb + fa1, 16);   \
    v16bf a2 = frag16(cb + fa2, 16), a3 = frag16(cb + fa3, 16);   \
    v16bf b0 = frag16(cb + fb0, 16), b1 = frag16(cb + fb1, 16);   \
    v16bf b2 = frag16(cb + fb2, 16), b3 = frag16(cb + fb3, 16);   \
    acc[0][0] = __builtin_amdgcn_wmma_f32_16x16x32_bf16(false, a0, false, b0, (short)0, acc[0][0], false, false); \
    acc[0][1] = __builtin_amdgcn_wmma_f32_16x16x32_bf16(false, a0, false, b1, (short)0, acc[0][1], false, false); \
    acc[0][2] = __builtin_amdgcn_wmma_f32_16x16x32_bf16(false, a0, false, b2, (short)0, acc[0][2], false, false); \
    acc[0][3] = __builtin_amdgcn_wmma_f32_16x16x32_bf16(false, a0, false, b3, (short)0, acc[0][3], false, false); \
    acc[1][0] = __builtin_amdgcn_wmma_f32_16x16x32_bf16(false, a1, false, b0, (short)0, acc[1][0], false, false); \
    acc[1][1] = __builtin_amdgcn_wmma_f32_16x16x32_bf16(false, a1, false, b1, (short)0, acc[1][1], false, false); \
    acc[1][2] = __builtin_amdgcn_wmma_f32_16x16x32_bf16(false, a1, false, b2, (short)0, acc[1][2], false, false); \
    acc[1][3] = __builtin_amdgcn_wmma_f32_16x16x32_bf16(false, a1, false, b3, (short)0, acc[1][3], false, false); \
    acc[2][0] = __builtin_amdgcn_wmma_f32_16x16x32_bf16(false, a2, false, b0, (short)0, acc[2][0], false, false); \
    acc[2][1] = __builtin_amdgcn_wmma_f32_16x16x32_bf16(false, a2, false, b1, (short)0, acc[2][1], false, false); \
    acc[2][2] = __builtin_amdgcn_wmma_f32_16x16x32_bf16(false, a2, false, b2, (short)0, acc[2][2], false, false); \
    acc[2][3] = __builtin_amdgcn_wmma_f32_16x16x32_bf16(false, a2, false, b3, (short)0, acc[2][3], false, false); \
    acc[3][0] = __builtin_amdgcn_wmma_f32_16x16x32_bf16(false, a3, false, b0, (short)0, acc[3][0], false, false); \
    acc[3][1] = __builtin_amdgcn_wmma_f32_16x16x32_bf16(false, a3, false, b1, (short)0, acc[3][1], false, false); \
    acc[3][2] = __builtin_amdgcn_wmma_f32_16x16x32_bf16(false, a3, false, b2, (short)0, acc[3][2], false, false); \
    acc[3][3] = __builtin_amdgcn_wmma_f32_16x16x32_bf16(false, a3, false, b3, (short)0, acc[3][3], false, false); \
    if (MORE) {                                                   \
      STORE_TILES(NXT);                                           \
      __syncthreads();                                            \
    }                                                             \
  } while (0)

  // prologue: tile 0 into buffer 0
  LOAD_TILES(0);
  STORE_TILES(0);
  __syncthreads();

  // steady state: 63 buffer pairs, all loads unconditional
  for (int k0 = 0; k0 < K_DIM - 2 * BK; k0 += 2 * BK) {
    GEMM_STEP(0, HALF_ELEMS, k0 + BK, 1);
    GEMM_STEP(HALF_ELEMS, 0, k0 + 2 * BK, 1);
  }
  // peeled tail: load last tile, then compute-only step
  GEMM_STEP(0, HALF_ELEMS, K_DIM - BK, 1);
  GEMM_STEP(HALF_ELEMS, 0, 0, 0);
#undef GEMM_STEP
#undef STORE_TILES
#undef LOAD_TILES

  // ---- epilogue: C/D layout — VGPR r: M = r + lhalf*8, N = l16 ----
  #pragma unroll
  for (int mt = 0; mt < 4; ++mt) {
    #pragma unroll
    for (int nt = 0; nt < 4; ++nt) {
      const int colL = wn * 64 + nt * 16 + l16;
      const float iw = sIw[colL];
      const int rowL0 = wm * 64 + mt * 16 + lhalf * 8;
      #pragma unroll
      for (int r = 0; r < 8; ++r) {
        const int rowL = rowL0 + r;
        out[(size_t)(m0 + rowL) * N_DIM + (n0 + colL)] =
            acc[mt][nt][r] * sIx[rowL] * iw;
      }
    }
  }
}

// ---------------------------------------------------------------------------
// FALLBACK GEMM (small ws): fused f32->bf16 conversion while staging.
// ---------------------------------------------------------------------------
__global__ __launch_bounds__(256, 2) void cos_gemm_fused_kernel(
    const float* __restrict__ A, const float* __restrict__ W,
    const float* __restrict__ invx, const float* __restrict__ invw,
    float* __restrict__ out)
{
  __shared__ unsigned short As[BM * LDK];
  __shared__ unsigned short Bs[BN * LDK];
  __shared__ float sIx[BM];
  __shared__ float sIw[BN];

  const int tid = threadIdx.x;
  const int m0 = blockIdx.y * BM;
  const int n0 = blockIdx.x * BN;

  sIx[tid] = invx[m0 + tid];
  if (tid < BN) sIw[tid] = invw[n0 + tid];

  const int wave  = tid >> 5;
  const int lane  = tid & 31;
  const int wm    = wave & 3;
  const int wn    = wave >> 2;
  const int lhalf = lane >> 4;
  const int l16   = lane & 15;

  v8f acc[4][4];
  #pragma unroll
  for (int i = 0; i < 4; ++i)
    #pragma unroll
    for (int j = 0; j < 4; ++j) acc[i][j] = v8f{};

  for (int k0 = 0; k0 < K_DIM; k0 += BK) {
    #pragma unroll
    for (int i = 0; i < 8; ++i) {
      const int slot = tid + i * 256;
      const int row  = slot >> 3;
      const int c4   = (slot & 7) << 2;
      float4 va = *(const float4*)(A + (size_t)(m0 + row) * K_DIM + k0 + c4);
      *(uint2*)&As[row * LDK + c4] = pk4_bf16(va);
    }
    #pragma unroll
    for (int i = 0; i < 4; ++i) {
      const int slot = tid + i * 256;
      const int row  = slot >> 3;
      const int c4   = (slot & 7) << 2;
      float4 vb = *(const float4*)(W + (size_t)(n0 + row) * K_DIM + k0 + c4);
      *(uint2*)&Bs[row * LDK + c4] = pk4_bf16(vb);
    }
    __syncthreads();

    v16bf af[4], bf[4];
    #pragma unroll
    for (int mt = 0; mt < 4; ++mt)
      af[mt] = frag16(&As[(wm * 64 + mt * 16 + l16) * LDK + (lhalf << 3)], 16);
    #pragma unroll
    for (int nt = 0; nt < 4; ++nt)
      bf[nt] = frag16(&Bs[(wn * 64 + nt * 16 + l16) * LDK + (lhalf << 3)], 16);

    #pragma unroll
    for (int mt = 0; mt < 4; ++mt)
      #pragma unroll
      for (int nt = 0; nt < 4; ++nt)
        acc[mt][nt] = __builtin_amdgcn_wmma_f32_16x16x32_bf16(
            false, af[mt], false, bf[nt],
            (short)0, acc[mt][nt], false, false);

    __syncthreads();
  }

  #pragma unroll
  for (int mt = 0; mt < 4; ++mt) {
    #pragma unroll
    for (int nt = 0; nt < 4; ++nt) {
      const int colL = wn * 64 + nt * 16 + l16;
      const float iw = sIw[colL];
      const int rowL0 = wm * 64 + mt * 16 + lhalf * 8;
      #pragma unroll
      for (int r = 0; r < 8; ++r) {
        const int rowL = rowL0 + r;
        out[(size_t)(m0 + rowL) * N_DIM + (n0 + colL)] =
            acc[mt][nt][r] * sIx[rowL] * iw;
      }
    }
  }
}

// ---------------------------------------------------------------------------
// Host-side launcher: path picked from ws_size (fixed per run -> deterministic)
// ---------------------------------------------------------------------------
extern "C" void kernel_launch(void* const* d_in, const int* in_sizes, int n_in,
                              void* d_out, int out_size, void* d_ws, size_t ws_size,
                              hipStream_t stream) {
  const float* x = (const float*)d_in[0];   // [4096, 4096]
  const float* w = (const float*)d_in[1];   // [8192, 4096]
  float* out = (float*)d_out;               // [4096, 8192]

  const size_t xb_elems = (size_t)M_DIM * K_DIM;
  const size_t wb_elems = (size_t)N_DIM * K_DIM;
  const size_t need_fast = (xb_elems + wb_elems) * sizeof(unsigned short) +
                           (size_t)(M_DIM + N_DIM) * sizeof(float);
  dim3 grid(N_DIM / BN, M_DIM / BM);        // (64, 16)

  if (ws_size >= need_fast) {
    unsigned short* xb = (unsigned short*)d_ws;
    unsigned short* wb = xb + xb_elems;
    float* invx = (float*)(wb + wb_elems);
    float* invw = invx + M_DIM;
    conv_norm_kernel<<<M_DIM, 256, 0, stream>>>(x, xb, invx, K_DIM);
    conv_norm_kernel<<<N_DIM, 256, 0, stream>>>(w, wb, invw, K_DIM);
    cos_gemm_bf16_kernel<<<grid, 256, 0, stream>>>(xb, wb, invx, invw, out);
  } else {
    float* invx = (float*)d_ws;             // 48 KB total
    float* invw = invx + M_DIM;
    inv_norm_kernel<<<M_DIM, 256, 0, stream>>>(x, invx, K_DIM);
    inv_norm_kernel<<<N_DIM, 256, 0, stream>>>(w, invw, K_DIM);
    cos_gemm_fused_kernel<<<grid, 256, 0, stream>>>(x, w, invx, invw, out);
  }
}
